// CustomHyperbolicLinear_55396488183986
// MI455X (gfx1250) — matrix operand order repaired and checked by reading, not compiled, not run on hardware
//
#include <hip/hip_runtime.h>
#include <hip/hip_fp16.h>

typedef __attribute__((ext_vector_type(16))) _Float16 v16h;
typedef __attribute__((ext_vector_type(8)))  float    v8f;

#define EPSN  1e-15f
#define MAXN  (1.0f - 1e-5f)
#define ATCL  (1.0f - 1e-7f)

__device__ __forceinline__ float fast_tanh(float x) {
#if __has_builtin(__builtin_amdgcn_tanhf)
    return __builtin_amdgcn_tanhf(x);   // gfx1250 v_tanh_f32
#else
    return tanhf(x);
#endif
}

__device__ __forceinline__ float fast_sqrt(float x) {
#if __has_builtin(__builtin_amdgcn_sqrtf)
    return __builtin_amdgcn_sqrtf(x);   // raw v_sqrt_f32 (no denorm fixup path)
#else
    return sqrtf(x);
#endif
}

__device__ __forceinline__ float fast_rcp(float x) {
#if __has_builtin(__builtin_amdgcn_rcpf)
    return __builtin_amdgcn_rcpf(x);    // raw v_rcp_f32 (no Newton refine)
#else
    return 1.0f / x;
#endif
}

__device__ __forceinline__ void cvt4(v16h& dst, int base, float4 f) {
    dst[base + 0] = (_Float16)f.x;
    dst[base + 1] = (_Float16)f.y;
    dst[base + 2] = (_Float16)f.z;
    dst[base + 3] = (_Float16)f.w;
}

__device__ __forceinline__ float sq4(float4 f) {
    return f.x * f.x + f.y * f.y + f.z * f.z + f.w * f.w;
}

// Folding butterfly over the 16-lane half: input S[0..7] per lane, output =
// sum over the 16 lanes of component (lane&7). Masks 4/2/1 halve the vector
// while pairing lanes whose target components differ; mask 8 completes the sum.
// 8 ds_bpermute instead of 32, and each lane ends with exactly its row's total.
__device__ __forceinline__ float fold_reduce8(const float S[8], int lane) {
    const bool b4 = (lane & 4) != 0;
    float k4[4], s4[4];
#pragma unroll
    for (int j = 0; j < 4; ++j) {
        k4[j] = b4 ? S[j + 4] : S[j];
        s4[j] = b4 ? S[j]     : S[j + 4];
    }
#pragma unroll
    for (int j = 0; j < 4; ++j) k4[j] += __shfl_xor(s4[j], 4, 32);
    const bool b2 = (lane & 2) != 0;
    float k2[2], s2[2];
#pragma unroll
    for (int j = 0; j < 2; ++j) {
        k2[j] = b2 ? k4[j + 2] : k4[j];
        s2[j] = b2 ? k4[j]     : k4[j + 2];
    }
#pragma unroll
    for (int j = 0; j < 2; ++j) k2[j] += __shfl_xor(s2[j], 2, 32);
    const bool b1 = (lane & 1) != 0;
    float k1 = b1 ? k2[1] : k2[0];
    float s1 = b1 ? k2[0] : k2[1];
    k1 += __shfl_xor(s1, 1, 32);
    k1 += __shfl_xor(k1, 8, 32);
    return k1;
}

// One wave computes a 16x64 output tile:
//   mx = x_tile(16x64) @ W^T(64x64) via 8x v_wmma_f32_16x16x32_f16,
//   then the hyperbolic chain using only per-row scalars (no output-side
//   norm reductions beyond ||mx||^2 and <mx,hb>).
__global__ __launch_bounds__(256)
void hyp_linear_wmma(const float* __restrict__ x,
                     const float* __restrict__ W,
                     const float* __restrict__ bias,
                     float* __restrict__ out,
                     int ntiles)
{
    const int lane = threadIdx.x & 31;
    const int wave = threadIdx.x >> 5;
    const int half = lane >> 4;      // 0: lanes 0-15, 1: lanes 16-31
    const int lm   = lane & 15;

    // ---------------- B fragments: W^T as f16, resident in VGPRs ----------------
    // Col tile t, K-chunk c: lane holds column j = 16t+lm,
    // K = 32c + 16*half .. +15  -> 16 contiguous floats of W row j.
    v16h bfrag[4][2];
#pragma unroll
    for (int t = 0; t < 4; ++t) {
        const float* wrow = W + (size_t)(16 * t + lm) * 64;
#pragma unroll
        for (int c = 0; c < 2; ++c) {
            const float* p = wrow + 32 * c + 16 * half;
#pragma unroll
            for (int q = 0; q < 4; ++q)
                cvt4(bfrag[t][c], 4 * q, *(const float4*)(p + 4 * q));
        }
    }

    // ---------------- hyperbolic bias (constant per wave) ----------------
    float hb[4];
    float y2;
    {
        float bv[4];
        float pb = 0.0f;
#pragma unroll
        for (int t = 0; t < 4; ++t) {
            bv[t] = bias[16 * t + lm];
            pb += bv[t] * bv[t];
        }
#pragma unroll
        for (int m = 1; m <= 8; m <<= 1)
            pb += __shfl_xor(pb, m, 32);
        const float bn  = fmaxf(fast_sqrt(pb), EPSN);
        const float tb  = fast_tanh(bn);              // ||expmap0(b)||
        const float sc0 = tb * fast_rcp(bn);
        const float hbn = fmaxf(tb, EPSN);
        const float s   = (hbn > MAXN) ? (MAXN * fast_rcp(hbn)) : 1.0f;
#pragma unroll
        for (int t = 0; t < 4; ++t) hb[t] = bv[t] * sc0 * s;
        const float yn = hbn * s;
        y2 = yn * yn;
    }

    const int gw = blockIdx.x * (blockDim.x >> 5) + wave;
    const int nw = gridDim.x * (blockDim.x >> 5);

    for (int tile = gw; tile < ntiles; tile += nw) {
        const int R = tile * 16;
        const float* xr = x + (size_t)(R + lm) * 64;

        if (tile + nw < ntiles)                        // global_prefetch_b8
            __builtin_prefetch(x + (size_t)((tile + nw) * 16 + lm) * 64, 0, 1);

        // ---- A fragments (f16) + per-lane ||x||^2 partial ----
        // A f16 k order used here (same as B): lanes<16 hold K = kb..kb+15;
        // lanes>=16 hold K = kb+16..kb+31 (kb = 32c).
        v16h afrag[2];
        float px = 0.0f;
#pragma unroll
        for (int c = 0; c < 2; ++c) {
            const float* p0 = xr + 32 * c + 16 * half;
            const float* p1 = xr + 32 * c + 8 + 16 * half;
            float4 a0 = *(const float4*)(p0);
            float4 a1 = *(const float4*)(p0 + 4);
            float4 a2 = *(const float4*)(p1);
            float4 a3 = *(const float4*)(p1 + 4);
            px += sq4(a0) + sq4(a1) + sq4(a2) + sq4(a3);
            cvt4(afrag[c], 0,  a0);
            cvt4(afrag[c], 4,  a1);
            cvt4(afrag[c], 8,  a2);
            cvt4(afrag[c], 12, a3);
        }
        px += __shfl_xor(px, 16, 32);                  // ||x_row(lm)||^2

        // artanh once per lane (its own row), redistributed later as needed
        float ainv_self;
        {
            const float xn = fmaxf(fast_sqrt(px), EPSN);
            const float z  = fminf(xn, ATCL);
            ainv_self = 0.5f * __logf((1.0f + z) * fast_rcp(1.0f - z)) * fast_rcp(xn);
        }

        // ---------------- GEMM: 8x v_wmma_f32_16x16x32_f16 ----------------
        v8f acc[4];
#pragma unroll
        for (int t = 0; t < 4; ++t) {
            v8f z0 = {0.f, 0.f, 0.f, 0.f, 0.f, 0.f, 0.f, 0.f};
            z0 = __builtin_amdgcn_wmma_f32_16x16x32_f16(
                    false, afrag[0], false, bfrag[t][0], (short)0, z0, false, false);
            acc[t] = __builtin_amdgcn_wmma_f32_16x16x32_f16(
                    false, afrag[1], false, bfrag[t][1], (short)0, z0, false, false);
        }

        // ---- per-row partials as v8f vector math (maps onto v_pk_* ops) ----
        v8f svv = acc[0] * acc[0];
        svv += acc[1] * acc[1];
        svv += acc[2] * acc[2];
        svv += acc[3] * acc[3];                         // Sum_j mx^2 partial
        v8f pvv = acc[0] * hb[0];
        pvv += acc[1] * hb[1];
        pvv += acc[2] * hb[2];
        pvv += acc[3] * hb[3];                          // <mx, hb> partial

        float Sa[8], Pa[8];
#pragma unroll
        for (int i = 0; i < 8; ++i) { Sa[i] = svv[i]; Pa[i] = pvv[i]; }

        // folding reduce: lane ends with totals for row ri = 8*half + (lane&7)
        const float sv_r = fold_reduce8(Sa, lane);
        const float pv_r = fold_reduce8(Pa, lane);
        const float ainv_r = __shfl(ainv_self, 8 * half + (lane & 7), 32);

        // ---- per-row scalar chain (ONCE per lane): matvec scale, proj,
        //      mobius_add coefficients, final proj (all analytic) ----
        float ca_self, cb_self;
        {
            const float mxn  = fmaxf(fast_sqrt(sv_r), EPSN);
            const float tn   = fast_tanh(mxn * ainv_r);
            const float resn = fmaxf(tn, EPSN);
            const float s1   = (resn > MAXN) ? (MAXN * fast_rcp(resn)) : 1.0f;
            const float f1   = tn * fast_rcp(mxn) * s1; // res = f1 * mx
            const float rn   = resn * s1;
            const float x2   = rn * rn;
            const float xy   = f1 * pv_r;
            const float den  = fmaxf(1.0f + 2.0f * xy + x2 * y2, EPSN);
            const float rden = fast_rcp(den);
            const float Ai   = (1.0f + 2.0f * xy + y2) * rden;
            const float Bi   = (1.0f - x2) * rden;
            const float on2  = Ai * Ai * x2 + 2.0f * Ai * Bi * xy + Bi * Bi * y2;
            const float on   = fmaxf(fast_sqrt(on2), EPSN);
            const float s2   = (on > MAXN) ? (MAXN * fast_rcp(on)) : 1.0f;
            ca_self = s2 * Ai * f1;                     // applies to mx
            cb_self = s2 * Bi;                          // applies to hb
        }

        // broadcast row coefficients back to the half (row 8*half+i at lane 16*half+i)
        v8f cav, cbv;
#pragma unroll
        for (int i = 0; i < 8; ++i) {
            cav[i] = __shfl(ca_self, 16 * half + i, 32);
            cbv[i] = __shfl(cb_self, 16 * half + i, 32);
        }

        // ---- epilogue as vector FMAs + stores (row = R+8*half+i, col = 16t+lm) ----
        float* obase = out + (size_t)(R + 8 * half) * 64 + lm;
#pragma unroll
        for (int t = 0; t < 4; ++t) {
            v8f o = cav * acc[t] + cbv * hb[t];
#pragma unroll
            for (int i = 0; i < 8; ++i)
                obase[(size_t)i * 64 + 16 * t] = o[i];
        }
    }
}

extern "C" void kernel_launch(void* const* d_in, const int* in_sizes, int n_in,
                              void* d_out, int out_size, void* d_ws, size_t ws_size,
                              hipStream_t stream) {
    const float* x = (const float*)d_in[0];   // [N, 64]
    const float* w = (const float*)d_in[1];   // [64, 64]
    const float* b = (const float*)d_in[2];   // [64]
    float* out = (float*)d_out;               // [N, 64] float32

    const int N      = in_sizes[0] / 64;
    const int ntiles = N / 16;                // N = 2^21 -> exact
    const int blocks = 1024;                  // 8192 waves, ~16 tiles each
    hyp_linear_wmma<<<blocks, 256, 0, stream>>>(x, w, b, out, ntiles);
}
